// Model_62405874811636
// MI455X (gfx1250) — matrix operand order repaired and checked by reading, not compiled, not run on hardware
//
#include <hip/hip_runtime.h>

#define VOCAB   50000
#define EMBED   128
#define CONTENT 200
#define HIDDEN  1024
#define LABELS  2
#define BATCH   4096
#define KDIM    (EMBED * CONTENT)   // 25600

#define LDA 136   // padded LDS row stride (elems): 272B = 68 banks -> 4-bank shift/row
#define LDB 136

typedef __attribute__((ext_vector_type(16))) __bf16 v16bf;
typedef __attribute__((ext_vector_type(8)))  float  v8f;

__device__ __forceinline__ unsigned f32_to_bf16(float f) {
    union { float f; unsigned u; } c; c.f = f;
    unsigned r = c.u + 0x7FFFu + ((c.u >> 16) & 1u);   // round-to-nearest-even
    return r >> 16;
}

// Async global -> LDS copy, 16B per lane. ASYNCcnt-tracked (CDNA5 ISA).
__device__ __forceinline__ void async_copy_b128(unsigned short* lds_dst,
                                                const unsigned short* gsrc) {
    unsigned loff = (unsigned)(uintptr_t)lds_dst;   // AS3 offset == low 32 bits
    asm volatile("global_load_async_to_lds_b128 %0, %1, off"
                 :: "v"(loff), "v"(gsrc) : "memory");
}
__device__ __forceinline__ void wait_asynccnt0() {
    asm volatile("s_wait_asynccnt 0" ::: "memory");
}

// 16B-aligned LDS fragment fetch -> ds_load_b128
__device__ __forceinline__ uint4 lds_b128(const unsigned short* p) {
    return *(const uint4*)__builtin_assume_aligned(p, 16);
}

// ---- Kernel 1: w1 f32 [H,K] -> bf16 [H,K] ----
__global__ __launch_bounds__(256) void w1_to_bf16_kernel(
        const float* __restrict__ w1, unsigned short* __restrict__ w1bf) {
    size_t i = ((size_t)blockIdx.x * 256 + threadIdx.x) * 4;
    float4 f = *(const float4*)(w1 + i);
    uint2 p;
    p.x = f32_to_bf16(f.x) | (f32_to_bf16(f.y) << 16);
    p.y = f32_to_bf16(f.z) | (f32_to_bf16(f.w) << 16);
    *(uint2*)(w1bf + i) = p;
}

// ---- Kernel 2: zero partial logits ----
__global__ __launch_bounds__(256) void zero_logits_kernel(float* __restrict__ logits) {
    int i = blockIdx.x * 256 + threadIdx.x;
    if (i < BATCH * LABELS) logits[i] = 0.0f;
}

// ---- Kernel 3: fused gather + bf16 WMMA GEMM1 + bias/ReLU + partial GEMM2 ----
struct Smem {
    union {
        unsigned short a[128 * LDA];   // A tile: 128 rows x 128 k (bf16, padded)
        float          h[128 * 64];    // h tile for epilogue (aliases A)
    };
    unsigned short b[64 * LDB];        // B tile: 64 cols x 128 k (bf16, padded)
};

__global__ __launch_bounds__(256) void gemm1_fused_kernel(
        const int*            __restrict__ x,       // [BATCH, CONTENT]
        const float*          __restrict__ emb,     // [VOCAB, EMBED]
        const unsigned short* __restrict__ w1bf,    // [HIDDEN, KDIM] bf16
        const float*          __restrict__ b1,      // [HIDDEN]
        const float*          __restrict__ w2,      // [LABELS, HIDDEN]
        float*                __restrict__ logits)  // [BATCH, LABELS] partial
{
    __shared__ Smem sm;

    const int tid  = threadIdx.x;
    const int lane = tid & 31;
    const int wave = tid >> 5;        // 0..7
    const int wm   = wave & 3;        // wave M position (4 x 32 rows)
    const int wn   = wave >> 2;       // wave N position (2 x 32 cols)
    const bool lo  = lane < 16;
    const int l16  = lane & 15;

    const int col0 = blockIdx.x * 64;     // hidden-unit base   (16 blocks)
    const int row0 = blockIdx.y * 128;    // batch-sample base  (32 blocks)

    // A-loader: thread -> (row, 64-element half of the 128-wide token row)
    const int a_row  = tid >> 1;          // 0..127
    const int a_half = (tid & 1) * 64;    // 0 or 64
    // B-loader: thread -> (n, 32-element quarter)
    const int b_n = tid >> 2;             // 0..63
    const int b_q = (tid & 3) * 32;       // 0,32,64,96

    const int* xrow = x + (size_t)(row0 + a_row) * CONTENT;

    v8f acc[2][2] = {};

    for (int tok = 0; tok < CONTENT; ++tok) {
        __syncthreads();   // all waves done reading previous tiles

        // ---- B tile: async bf16 w1 -> LDS (4 x 16B per lane, ASYNCcnt) ----
        {
            const unsigned short* bsrc =
                w1bf + (size_t)(col0 + b_n) * KDIM + tok * EMBED + b_q;
            unsigned short* bdst = &sm.b[b_n * LDB + b_q];
            #pragma unroll
            for (int j = 0; j < 4; ++j)
                async_copy_b128(bdst + j * 8, bsrc + j * 8);
        }
        // ---- A tile: gather one token row, convert f32 -> bf16 ----
        {
            const int id = xrow[tok];
            unsigned short* dst = &sm.a[a_row * LDA + a_half];
            if (id == 0) {                           // padding_idx = 0
                uint4 z = {0u, 0u, 0u, 0u};
                #pragma unroll
                for (int i = 0; i < 64; i += 8) *(uint4*)(dst + i) = z;
            } else {
                const float* src = emb + (size_t)id * EMBED + a_half;
                #pragma unroll
                for (int i = 0; i < 64; i += 8) {
                    float4 f0 = *(const float4*)(src + i);
                    float4 f1 = *(const float4*)(src + i + 4);
                    uint4 p;
                    p.x = f32_to_bf16(f0.x) | (f32_to_bf16(f0.y) << 16);
                    p.y = f32_to_bf16(f0.z) | (f32_to_bf16(f0.w) << 16);
                    p.z = f32_to_bf16(f1.x) | (f32_to_bf16(f1.y) << 16);
                    p.w = f32_to_bf16(f1.z) | (f32_to_bf16(f1.w) << 16);
                    *(uint4*)(dst + i) = p;
                }
            }
        }
        wait_asynccnt0();   // our async B writes have landed in LDS
        __syncthreads();    // (compiler also drains DScnt before the barrier)

        // ---- 4 K-steps of 32 -> 16 WMMAs per stage ----
        #pragma unroll
        for (int ks = 0; ks < 4; ++ks) {
            const int kofs = ks * 32;
            union Frag { uint4 q[2]; v16bf v; };
            Frag afr[2];
            #pragma unroll
            for (int mi = 0; mi < 2; ++mi) {
                const unsigned short* pa =
                    &sm.a[(wm * 32 + mi * 16 + l16) * LDA + kofs + (lo ? 0 : 8)];
                afr[mi].q[0] = lds_b128(pa);        // K 0-7   / 8-15
                afr[mi].q[1] = lds_b128(pa + 16);   // K 16-23 / 24-31
            }
            Frag bfr[2];
            #pragma unroll
            for (int ni = 0; ni < 2; ++ni) {
                const unsigned short* pb =
                    &sm.b[(wn * 32 + ni * 16 + l16) * LDB + kofs + (lo ? 0 : 8)];
                bfr[ni].q[0] = lds_b128(pb);        // K 0-7   / 8-15
                bfr[ni].q[1] = lds_b128(pb + 16);   // K 16-23 / 24-31
            }
            #pragma unroll
            for (int mi = 0; mi < 2; ++mi)
                #pragma unroll
                for (int ni = 0; ni < 2; ++ni)
                    acc[mi][ni] = __builtin_amdgcn_wmma_f32_16x16x32_bf16(
                        false, afr[mi].v, false, bfr[ni].v,
                        (short)0, acc[mi][ni], false, false);
        }
    }

    // ---- epilogue: +b1, ReLU, stash h tile (aliases sm.a) ----
    __syncthreads();
    #pragma unroll
    for (int mi = 0; mi < 2; ++mi) {
        #pragma unroll
        for (int ni = 0; ni < 2; ++ni) {
            const int n_loc = wn * 32 + ni * 16 + l16;
            const float bias = b1[col0 + n_loc];
            #pragma unroll
            for (int j = 0; j < 8; ++j) {
                const int m_loc = wm * 32 + mi * 16 + (lo ? j : j + 8);
                float v = acc[mi][ni][j] + bias;
                sm.h[m_loc * 64 + n_loc] = v > 0.0f ? v : 0.0f;
            }
        }
    }
    __syncthreads();

    // ---- partial GEMM2: logits[row, lab] += sum_n h[row,n] * w2[lab, col0+n] ----
    {
        const int r   = tid >> 1;     // 0..127
        const int lab = tid & 1;      // 0..1
        const float* w2row = w2 + lab * HIDDEN + col0;
        float s = 0.0f;
        #pragma unroll 8
        for (int n = 0; n < 64; ++n)
            s += sm.h[r * 64 + n] * w2row[n];
        atomicAdd(&logits[(row0 + r) * LABELS + lab], s);
    }
}

// ---- Kernel 4: +b2, 2-way softmax -> out ----
__global__ __launch_bounds__(256) void finalize_kernel(
        const float* __restrict__ logits, const float* __restrict__ b2,
        float* __restrict__ out) {
    int s = blockIdx.x * 256 + threadIdx.x;
    if (s >= BATCH) return;
    float l0 = logits[2 * s + 0] + b2[0];
    float l1 = logits[2 * s + 1] + b2[1];
    float m  = fmaxf(l0, l1);
    float e0 = __expf(l0 - m);
    float e1 = __expf(l1 - m);
    float inv = 1.0f / (e0 + e1);
    out[2 * s + 0] = e0 * inv;
    out[2 * s + 1] = e1 * inv;
}

extern "C" void kernel_launch(void* const* d_in, const int* in_sizes, int n_in,
                              void* d_out, int out_size, void* d_ws, size_t ws_size,
                              hipStream_t stream) {
    const int*   x   = (const int*)d_in[0];
    const float* emb = (const float*)d_in[1];
    const float* w1  = (const float*)d_in[2];
    const float* b1  = (const float*)d_in[3];
    const float* w2  = (const float*)d_in[4];
    const float* b2  = (const float*)d_in[5];
    float* out = (float*)d_out;

    unsigned short* w1bf = (unsigned short*)d_ws;                       // 52.4 MB
    float* logits = (float*)((char*)d_ws + (size_t)HIDDEN * KDIM * 2);  // 32 KB

    w1_to_bf16_kernel<<<dim3((HIDDEN * KDIM) / 1024), 256, 0, stream>>>(w1, w1bf);
    zero_logits_kernel<<<dim3((BATCH * LABELS + 255) / 256), 256, 0, stream>>>(logits);
    gemm1_fused_kernel<<<dim3(HIDDEN / 64, BATCH / 128), 256, 0, stream>>>(
        x, emb, w1bf, b1, w2, logits);
    finalize_kernel<<<dim3(BATCH / 256), 256, 0, stream>>>(logits, b2, out);
}
